// Transformer_43215960932676
// MI455X (gfx1250) — matrix operand order repaired and checked by reading, not compiled, not run on hardware
//
#include <hip/hip_runtime.h>

#define SEQ 2048
#define BB 2
#define DD 512
#define HH 8
#define DHH 64
#define FF 2048
#define LL 4
#define MROWS (SEQ * BB) /* 4096 */
#define TLDK 40          /* LDS row stride in halfs (32 + 8 pad) */

typedef __attribute__((ext_vector_type(16))) __bf16 v16bf;
typedef __attribute__((ext_vector_type(16))) unsigned short v16us;
typedef __attribute__((ext_vector_type(8))) unsigned short v8us;
typedef __attribute__((ext_vector_type(8))) float v8f;

__device__ __forceinline__ unsigned short f2bf(float f) {
  unsigned int u = __builtin_bit_cast(unsigned int, f);
  u += 0x7fffu + ((u >> 16) & 1u);  // round-to-nearest-even
  return (unsigned short)(u >> 16);
}

__device__ __forceinline__ v8f zero8() {
  v8f z;
#pragma unroll
  for (int i = 0; i < 8; i++) z[i] = 0.0f;
  return z;
}

__device__ __forceinline__ v8f wmma_bf16(v16us a, v16us b, v8f c) {
  return __builtin_amdgcn_wmma_f32_16x16x32_bf16(
      false, __builtin_bit_cast(v16bf, a), false, __builtin_bit_cast(v16bf, b),
      (short)0, c, false, false);
}

// C[M,N] = act(A[M,K] @ W[N,K]^T + bias). Logical rows of A/C are batch-major
// (m = b*SEQ + s). a_smajor=1 means the A buffer is seq-major [SEQ,B,K]
// (phys row = s*B + b). C is always written batch-major, ld = N.
__global__ __launch_bounds__(256) void gemm_xwT(
    const float* __restrict__ A, const float* __restrict__ W,
    const float* __restrict__ bias, float* __restrict__ C, int M, int N, int K,
    int a_smajor, int relu) {
  __shared__ __align__(16) unsigned short lA[128 * TLDK];
  __shared__ __align__(16) unsigned short lB[128 * TLDK];
  const int tid = threadIdx.x;
  const int lane = tid & 31;
  const int wid = tid >> 5;
  const int wm = (wid & 3) * 32;   // wave M block within tile
  const int wn = (wid >> 2) * 64;  // wave N block within tile
  const int m0 = blockIdx.y * 128;
  const int n0 = blockIdx.x * 128;
  const int lo = lane & 15;

  v8f acc[2][4];
#pragma unroll
  for (int i = 0; i < 2; i++)
#pragma unroll
    for (int j = 0; j < 4; j++) acc[i][j] = zero8();

  for (int kk = 0; kk < K; kk += 32) {
    __syncthreads();
#pragma unroll
    for (int p = 0; p < 4; p++) {
      int idx = tid + p * 256;
      int r = idx >> 3;
      int c4 = (idx & 7) << 2;
      int mlog = m0 + r;
      int mphys = a_smajor ? ((mlog & (SEQ - 1)) * BB + (mlog >> 11)) : mlog;
      float4 va = *(const float4*)(A + (size_t)mphys * K + kk + c4);
      unsigned short* pa = lA + r * TLDK + c4;
      pa[0] = f2bf(va.x); pa[1] = f2bf(va.y); pa[2] = f2bf(va.z); pa[3] = f2bf(va.w);
      float4 vb = *(const float4*)(W + (size_t)(n0 + r) * K + kk + c4);
      unsigned short* pb = lB + r * TLDK + c4;
      pb[0] = f2bf(vb.x); pb[1] = f2bf(vb.y); pb[2] = f2bf(vb.z); pb[3] = f2bf(vb.w);
    }
    __syncthreads();

    // A fragment: lane<16 holds K 0..7 & 16..23, lane>=16 holds 8..15 & 24..31
    v16us af[2], bf[4];
    const int koff8 = (lane < 16) ? 0 : 8;
#pragma unroll
    for (int sm = 0; sm < 2; sm++) {
      const unsigned short* p = lA + (wm + sm * 16 + lo) * TLDK + koff8;
      v8us x = *(const v8us*)p;
      v8us y = *(const v8us*)(p + 16);
      af[sm] = __builtin_shufflevector(x, y, 0, 1, 2, 3, 4, 5, 6, 7, 8, 9, 10,
                                       11, 12, 13, 14, 15);
    }
    // B fragment: lane<16 holds K 0..7 & 16..23, lane>=16 holds 8..15 & 24..31
    const int koff16 = (lane < 16) ? 0 : 16;
#pragma unroll
    for (int sn = 0; sn < 4; sn++) {
      const unsigned short* p = lB + (wn + sn * 16 + lo) * TLDK + koff8;
      v8us x = *(const v8us*)p;
      v8us y = *(const v8us*)(p + 16);
      bf[sn] = __builtin_shufflevector(x, y, 0, 1, 2, 3, 4, 5, 6, 7, 8, 9, 10,
                                       11, 12, 13, 14, 15);
    }
#pragma unroll
    for (int sm = 0; sm < 2; sm++)
#pragma unroll
      for (int sn = 0; sn < 4; sn++)
        acc[sm][sn] = wmma_bf16(af[sm], bf[sn], acc[sm][sn]);
  }

  const int hi8 = (lane >> 4) * 8;
#pragma unroll
  for (int sm = 0; sm < 2; sm++)
#pragma unroll
    for (int sn = 0; sn < 4; sn++) {
      int col = n0 + wn + sn * 16 + lo;
      float bv = bias ? bias[col] : 0.0f;
#pragma unroll
      for (int r = 0; r < 8; r++) {
        int row = m0 + wm + sm * 16 + r + hi8;
        float v = acc[sm][sn][r] + bv;
        if (relu) v = fmaxf(v, 0.0f);
        C[(size_t)row * N + col] = v;
      }
    }
}

// Flash attention, one wave per 16-query tile per (b,h).
// Scores computed transposed (S^T = K*Q^T) so P lands in the A-operand layout.
__global__ __launch_bounds__(32) void flash_attn(
    const float* __restrict__ Q, const float* __restrict__ K,
    const float* __restrict__ V, float* __restrict__ O, int ldq, int ldk,
    int ldv, int ldo, long bsq, long hsq, long bsk, long hsk, long bsv,
    long hsv, long bso, long hso, int nk, int causal) {
  const int lane = threadIdx.x;
  const int lo = lane & 15;
  const int hi = lane >> 4;
  const int bh = blockIdx.y;
  const int b = bh / HH, h = bh % HH;
  const float* qbp = Q + b * bsq + h * hsq;
  const float* kbp = K + b * bsk + h * hsk;
  const float* vbp = V + b * bsv + h * hsv;
  float* obp = O + b * bso + h * hso;
  const int qt = blockIdx.x * 16;
  const int koff16 = hi ? 16 : 0;
  const int koff8 = hi ? 8 : 0;

  // Q fragments (B operand): lane = query column, 16 contiguous d values.
  v16us qf[2];
  {
    const float* qr = qbp + (size_t)(qt + lo) * ldq;
#pragma unroll
    for (int dc = 0; dc < 2; dc++) {
      const float* p = qr + dc * 32 + koff16;
      v16us t;
#pragma unroll
      for (int e = 0; e < 16; e++) t[e] = f2bf(p[e]);
      qf[dc] = t;
    }
  }

  v8f oacc[4];
#pragma unroll
  for (int nb = 0; nb < 4; nb++) oacc[nb] = zero8();
  float mrun = -3.0e38f, lrun = 0.0f;
  const float sc = 0.125f * 1.44269504088896f;  // 1/sqrt(64) * log2(e)
  const int kend = causal ? (qt + 16) : nk;

  for (int kbase = 0; kbase < kend; kbase += 32) {
    // Two S^T tiles: M = key row (in VGPRs), N = query (in lanes).
    v8f s[2];
    s[0] = zero8();
    s[1] = zero8();
#pragma unroll
    for (int sub = 0; sub < 2; sub++) {
      const float* kr = kbp + (size_t)(kbase + sub * 16 + lo) * ldk;
#pragma unroll
      for (int dc = 0; dc < 2; dc++) {
        const float* p = kr + dc * 32;
        v16us kf;
#pragma unroll
        for (int e = 0; e < 8; e++) {
          kf[e] = f2bf(p[koff16 + e]);
          kf[8 + e] = f2bf(p[koff16 + 8 + e]);
        }
        s[sub] = wmma_bf16(kf, qf[dc], s[sub]);
      }
    }

    // Online softmax (per-lane query = qt+lo; lanes L and L^16 mirror stats).
    const int qg = qt + lo;
    float t0[8], t1[8];
    float mx = mrun;
#pragma unroll
    for (int r = 0; r < 8; r++) {
      int kg = kbase + r + hi * 8;
      float a0 = s[0][r] * sc;
      float a1 = s[1][r] * sc;
      if (causal) {
        if (kg > qg) a0 = -1.0e30f;
        if (kg + 16 > qg) a1 = -1.0e30f;
      }
      t0[r] = a0;
      t1[r] = a1;
      mx = fmaxf(mx, fmaxf(a0, a1));
    }
    mx = fmaxf(mx, __shfl_xor(mx, 16));
    float alpha = exp2f(mrun - mx);
    mrun = mx;
    float rs = 0.0f;
    v16us pf;  // P in A-operand layout: elems 0..7 tile0, 8..15 tile1
#pragma unroll
    for (int r = 0; r < 8; r++) {
      float p0 = exp2f(t0[r] - mx);
      float p1 = exp2f(t1[r] - mx);
      rs += p0 + p1;
      pf[r] = f2bf(p0);
      pf[8 + r] = f2bf(p1);
    }
    rs += __shfl_xor(rs, 16);
    lrun = lrun * alpha + rs;
    // Rescale O rows (row q = r + 8*hi) by alpha of that query.
#pragma unroll
    for (int r = 0; r < 8; r++) {
      float ar = __shfl(alpha, hi ? (r + 8) : r);
#pragma unroll
      for (int nb = 0; nb < 4; nb++) oacc[nb][r] *= ar;
    }
    // O += P @ V
#pragma unroll
    for (int nb = 0; nb < 4; nb++) {
      const int dcol = nb * 16 + lo;
      v16us vf;
#pragma unroll
      for (int e = 0; e < 16; e++)
        vf[e] = f2bf(vbp[(size_t)(kbase + koff8 + (e & 7) + ((e >> 3) << 4)) * ldv + dcol]);
      oacc[nb] = wmma_bf16(pf, vf, oacc[nb]);
    }
  }

  float linv = 1.0f / lrun;
#pragma unroll
  for (int r = 0; r < 8; r++) {
    float li = __shfl(linv, hi ? (r + 8) : r);
    int qrow = qt + r + hi * 8;
#pragma unroll
    for (int nb = 0; nb < 4; nb++)
      obp[(size_t)qrow * ldo + nb * 16 + lo] = oacc[nb][r] * li;
  }
}

// out = LayerNorm(x + y) * g + b. x/out seq-major rows (row = s*B + b);
// y_bmajor selects batch-major row mapping for y.
__global__ __launch_bounds__(128) void ln_res(const float* __restrict__ x,
                                              const float* __restrict__ y,
                                              int y_bmajor,
                                              const float* __restrict__ g,
                                              const float* __restrict__ bta,
                                              float* __restrict__ out) {
  __shared__ float red[128];
  const int row = blockIdx.x;
  const int s = row >> 1, b = row & 1;  // B == 2
  const int yrow = y_bmajor ? (b * SEQ + s) : row;
  const int tid = threadIdx.x;
  float v[4];
  float sum = 0.0f;
#pragma unroll
  for (int j = 0; j < 4; j++) {
    int c = tid * 4 + j;
    v[j] = x[(size_t)row * DD + c] + y[(size_t)yrow * DD + c];
    sum += v[j];
  }
  red[tid] = sum;
  __syncthreads();
  for (int off = 64; off > 0; off >>= 1) {
    if (tid < off) red[tid] += red[tid + off];
    __syncthreads();
  }
  float mu = red[0] * (1.0f / DD);
  __syncthreads();
  float sq = 0.0f;
#pragma unroll
  for (int j = 0; j < 4; j++) {
    float d = v[j] - mu;
    sq += d * d;
  }
  red[tid] = sq;
  __syncthreads();
  for (int off = 64; off > 0; off >>= 1) {
    if (tid < off) red[tid] += red[tid + off];
    __syncthreads();
  }
  float rstd = rsqrtf(red[0] * (1.0f / DD) + 1e-5f);
#pragma unroll
  for (int j = 0; j < 4; j++) {
    int c = tid * 4 + j;
    out[(size_t)row * DD + c] = (v[j] - mu) * rstd * g[c] + bta[c];
  }
}

extern "C" void kernel_launch(void* const* d_in, const int* in_sizes, int n_in,
                              void* d_out, int out_size, void* d_ws,
                              size_t ws_size, hipStream_t stream) {
  (void)in_sizes; (void)n_in; (void)out_size; (void)ws_size;
  const float* tgt = (const float*)d_in[0];
  const float* mem = (const float*)d_in[1];
  const float* qkv_w = (const float*)d_in[2];
  const float* sa_out_w = (const float*)d_in[3];
  const float* in_proj_w = (const float*)d_in[4];
  const float* in_proj_b = (const float*)d_in[5];
  const float* mha_out_w = (const float*)d_in[6];
  const float* mha_out_b = (const float*)d_in[7];
  const float* lin1_w = (const float*)d_in[8];
  const float* lin1_b = (const float*)d_in[9];
  const float* lin2_w = (const float*)d_in[10];
  const float* lin2_b = (const float*)d_in[11];
  const float* ln1_g = (const float*)d_in[12];
  const float* ln1_b = (const float*)d_in[13];
  const float* ln2_g = (const float*)d_in[14];
  const float* ln2_b = (const float*)d_in[15];
  const float* ln3_g = (const float*)d_in[16];
  const float* ln3_b = (const float*)d_in[17];

  float* ws = (float*)d_ws;
  const size_t NT = (size_t)MROWS * DD;  // 2M floats
  float* xb = ws;            // x, seq-major [S,B,D]
  float* qb = xb + NT;       // batch-major [B*S, D]
  float* kb = qb + NT;
  float* vb = kb + NT;
  float* ob = vb + NT;       // attention output, batch-major
  float* yb = ob + NT;       // sublayer output, batch-major
  float* hb = yb + NT;       // FFN hidden, [B*S, F]

  hipMemcpyAsync(xb, tgt, NT * sizeof(float), hipMemcpyDeviceToDevice, stream);

  dim3 blkG(256), blkA(32), blkL(128);
  dim3 g512(DD / 128, MROWS / 128);
  dim3 g2048(FF / 128, MROWS / 128);
  dim3 gAttn(SEQ / 16, BB * HH);

  const long SD = (long)SEQ * DD;   // batch stride
  const long HS = (long)SEQ * DHH;  // "plain-reshape" head stride

  for (int i = 0; i < LL; i++) {
    const float* qkv_i = qkv_w + (size_t)i * 3 * DD * DD;
    const float* ip_w = in_proj_w + (size_t)i * 3 * DD * DD;
    const float* ip_b = in_proj_b + (size_t)i * 3 * DD;
    // --- self attention (fused QKV, plain-reshape head split, causal) ---
    gemm_xwT<<<g512, blkG, 0, stream>>>(xb, qkv_i, nullptr, qb, MROWS, DD, DD, 1, 0);
    gemm_xwT<<<g512, blkG, 0, stream>>>(xb, qkv_i + DD * DD, nullptr, kb, MROWS, DD, DD, 1, 0);
    gemm_xwT<<<g512, blkG, 0, stream>>>(xb, qkv_i + 2 * DD * DD, nullptr, vb, MROWS, DD, DD, 1, 0);
    flash_attn<<<gAttn, blkA, 0, stream>>>(qb, kb, vb, ob, DHH, DHH, DHH, DHH,
                                           SD, HS, SD, HS, SD, HS, SD, HS, SEQ, 1);
    gemm_xwT<<<g512, blkG, 0, stream>>>(ob, sa_out_w + (size_t)i * DD * DD,
                                        nullptr, yb, MROWS, DD, DD, 0, 0);
    ln_res<<<MROWS, blkL, 0, stream>>>(xb, yb, 1, ln1_g + i * DD, ln1_b + i * DD, xb);
    // --- cross attention (torch MHA layout) ---
    gemm_xwT<<<g512, blkG, 0, stream>>>(xb, ip_w, ip_b, qb, MROWS, DD, DD, 1, 0);
    gemm_xwT<<<g512, blkG, 0, stream>>>(mem, ip_w + DD * DD, ip_b + DD, kb, MROWS, DD, DD, 1, 0);
    gemm_xwT<<<g512, blkG, 0, stream>>>(mem, ip_w + 2 * DD * DD, ip_b + 2 * DD, vb, MROWS, DD, DD, 1, 0);
    flash_attn<<<gAttn, blkA, 0, stream>>>(qb, kb, vb, ob, DD, DD, DD, DD, SD,
                                           (long)DHH, SD, (long)DHH, SD,
                                           (long)DHH, SD, (long)DHH, SEQ, 0);
    gemm_xwT<<<g512, blkG, 0, stream>>>(ob, mha_out_w + (size_t)i * DD * DD,
                                        mha_out_b + i * DD, yb, MROWS, DD, DD, 0, 0);
    ln_res<<<MROWS, blkL, 0, stream>>>(xb, yb, 1, ln2_g + i * DD, ln2_b + i * DD, xb);
    // --- feedforward ---
    gemm_xwT<<<g2048, blkG, 0, stream>>>(xb, lin1_w + (size_t)i * FF * DD,
                                         lin1_b + i * FF, hb, MROWS, FF, DD, 1, 1);
    gemm_xwT<<<g512, blkG, 0, stream>>>(hb, lin2_w + (size_t)i * DD * FF,
                                        lin2_b + i * DD, yb, MROWS, DD, FF, 0, 0);
    float* lnout = (i == LL - 1) ? (float*)d_out : xb;
    ln_res<<<MROWS, blkL, 0, stream>>>(xb, yb, 1, ln3_g + i * DD, ln3_b + i * DD, lnout);
  }
}
